// Self_Attention_66090956751110
// MI455X (gfx1250) — matrix operand order repaired and checked by reading, not compiled, not run on hardware
//
#include <hip/hip_runtime.h>
#include <hip/hip_bf16.h>

// Self-attention: x[4,4096,1024]f32, Wq/Wk/Wv[1024,512]f32 -> out[4,4096,512]f32
// Pipeline: (1) weights -> f16 transposed, (2) QKV projection via WMMA f16->f32
//           (16x64 strip per wave, 1/sqrt(dk) folded into Q), (3) flash attention
//           with 32-query tiles, WMMA f16->f32, online softmax, L2 prefetch.
// Workspace layout (f16): Qh[16384*512] | Kh[16384*512] | Vt[512*16384] | Wt[3*512*1024]
// => ~53.5 MB required in d_ws.

typedef __attribute__((ext_vector_type(16))) _Float16 v16h;
typedef __attribute__((ext_vector_type(8)))  _Float16 v8h;
typedef __attribute__((ext_vector_type(8)))  float    v8f;

union F16Frag { v16h v; v8h h8[2]; _Float16 h[16]; };

__device__ __forceinline__ v8f wmma_f16f32(const F16Frag& a, const F16Frag& b, v8f c) {
  // 8 args: (neg_a, A, neg_b, B, c_mod, C, reuse_a, reuse_b)
  return __builtin_amdgcn_wmma_f32_16x16x32_f16(false, a.v, false, b.v, (short)0, c, false, false);
}

#define N_ROWS 16384   // 4 * 4096 flattened rows
#define D_IN   1024
#define D_K    512
#define SEQ    4096
#define QSCALE 0.04419417382415922f   // 1/sqrt(512), folded into Q

// ---------------------------------------------------------------------------
// Kernel 1: Wq/Wk/Wv (f32 [1024,512] row-major) -> Wt (f16 [mat][512][1024])
// Wt[mat][n][k] = W[k][n]: WMMA B-fragments become two contiguous 16B runs per lane.
// ---------------------------------------------------------------------------
__global__ void __launch_bounds__(256) wconv_kernel(const float* __restrict__ Wq,
                                                    const float* __restrict__ Wk,
                                                    const float* __restrict__ Wv,
                                                    _Float16* __restrict__ Wt) {
  const int mat = blockIdx.y;
  const float* W = (mat == 0) ? Wq : (mat == 1) ? Wk : Wv;
  const int e = blockIdx.x * 256 + threadIdx.x;      // e = n*1024 + k
  const int n = e >> 10;
  const int k = e & 1023;
  Wt[(size_t)mat * (D_K * D_IN) + e] = (_Float16)W[k * D_K + n];
}

// ---------------------------------------------------------------------------
// Kernel 2: QKV projection. One wave per 16x64 output strip (A reused over 4
// N-tiles). K-loop 1024 in chunks of 32 -> v_wmma_f32_16x16x32_f16.
// Q written *pre-scaled* row-major f16; K row-major; V transposed Vt[dv][row].
// ---------------------------------------------------------------------------
__global__ void __launch_bounds__(128) qkv_proj_kernel(const float* __restrict__ x,
                                                       const _Float16* __restrict__ Wt,
                                                       _Float16* __restrict__ Qh,
                                                       _Float16* __restrict__ Kh,
                                                       _Float16* __restrict__ Vt) {
  const int wid  = threadIdx.x >> 5;
  const int lane = threadIdx.x & 31;
  const int h    = lane >> 4;          // lane half (fragment layout selector)
  const int ln   = lane & 15;

  const int gid = blockIdx.x * 4 + wid;      // 3 * 1024 * 8 = 24576 strips
  const int mat = gid >> 13;                 // 8192 strips per matrix
  const int rem = gid & 8191;
  const int mbase = (rem >> 3) << 4;         // 1024 row tiles
  const int nb0   = (rem & 7) << 6;          // 8 strips of 64 cols

  const float*    xrow  = x  + (size_t)(mbase + ln) * D_IN;
  const _Float16* wbase = Wt + (size_t)mat * (D_K * D_IN);

  v8f acc[4];
#pragma unroll
  for (int n = 0; n < 4; ++n)
#pragma unroll
    for (int r = 0; r < 8; ++r) acc[n][r] = 0.f;

  for (int kb = 0; kb < D_IN; kb += 32) {
    // A fragment (x, f32 -> f16): lanes 0-15 need K = kb+{0..7, 16..23},
    // lanes 16-31 need K = kb+{8..15, 24..31} => two contiguous 8-float runs.
    const float* p0 = xrow + kb + 8 * h;
    const float* p1 = p0 + 16;
    float4 f0 = *(const float4*)p0;
    float4 f1 = *(const float4*)(p0 + 4);
    float4 f2 = *(const float4*)p1;
    float4 f3 = *(const float4*)(p1 + 4);
    float t[16] = {f0.x, f0.y, f0.z, f0.w, f1.x, f1.y, f1.z, f1.w,
                   f2.x, f2.y, f2.z, f2.w, f3.x, f3.y, f3.z, f3.w};
    F16Frag a;
#pragma unroll
    for (int i = 0; i < 16; ++i) a.h[i] = (_Float16)t[i];

#pragma unroll
    for (int nt = 0; nt < 4; ++nt) {
      // B fragment (Wt row n): 8 f16 at K = kb + 8h and 8 at K = kb + 16 + 8h.
      F16Frag bfr;
      const v8h* bp = (const v8h*)(wbase + (size_t)(nb0 + nt * 16 + ln) * D_IN + kb + 8 * h);
      bfr.h8[0] = bp[0];
      bfr.h8[1] = bp[2];
      acc[nt] = wmma_f16f32(a, bfr, acc[nt]);
    }
  }

  // C layout: VGPR r holds (M = r + 8h, N = ln).
  if (mat < 2) {
    _Float16* dst = (mat == 0) ? Qh : Kh;
    const float s = (mat == 0) ? QSCALE : 1.f;
#pragma unroll
    for (int nt = 0; nt < 4; ++nt)
#pragma unroll
      for (int r = 0; r < 8; ++r) {
        const int m = mbase + 8 * h + r;
        dst[(size_t)m * D_K + nb0 + nt * 16 + ln] = (_Float16)(acc[nt][r] * s);
      }
  } else {
    // Transposed store: rows vary with r -> 8 contiguous f16 -> one b128 store.
#pragma unroll
    for (int nt = 0; nt < 4; ++nt) {
      v8h pk;
#pragma unroll
      for (int r = 0; r < 8; ++r) pk[r] = (_Float16)acc[nt][r];
      *(v8h*)&Vt[(size_t)(nb0 + nt * 16 + ln) * N_ROWS + mbase + 8 * h] = pk;
    }
  }
}

// ---------------------------------------------------------------------------
// Kernel 3: flash attention, 32-query tiles. One block per (batch, 32 queries),
// 8 waves. Wave j owns dk-slice [64j,64j+64) for scores and dv-slice for PV.
// K fragments reused across both query row-tiles; V fragments reused across
// both accumulator groups. Deterministic LDS reduce; prefetch of next tile.
// ---------------------------------------------------------------------------
__global__ void __launch_bounds__(256) attn_kernel(const _Float16* __restrict__ Qh,
                                                   const _Float16* __restrict__ Kh,
                                                   const _Float16* __restrict__ Vt,
                                                   float* __restrict__ out) {
  __shared__ float S_part[8][1024];   // per-wave partials: 2 row-tiles x 16x32
  __shared__ float S_red[1024];       // reduced scores
  __shared__ float bc[8][2][16];      // per-wave row-broadcast scratch

  const int b     = blockIdx.y;
  const int qbase = b * SEQ + blockIdx.x * 32;     // global query row base
  const int wid   = threadIdx.x >> 5;
  const int lane  = threadIdx.x & 31;
  const int h     = lane >> 4;
  const int ln    = lane & 15;
  const int dkb   = wid * 64;                      // this wave's dk / dv slice

  // Q fragments: 2 row-tiles x 2 k-chunks of the wave's 64-wide dk slice.
  F16Frag qf[2][2];
#pragma unroll
  for (int g = 0; g < 2; ++g)
#pragma unroll
    for (int c = 0; c < 2; ++c) {
      const _Float16* qp = Qh + (size_t)(qbase + g * 16 + ln) * D_K + dkb + c * 32;
      qf[g][c].h8[0] = *(const v8h*)(qp + 8 * h);
      qf[g][c].h8[1] = *(const v8h*)(qp + 16 + 8 * h);
    }

  v8f acc[2][4];
#pragma unroll
  for (int g = 0; g < 2; ++g)
#pragma unroll
    for (int n = 0; n < 4; ++n)
#pragma unroll
      for (int r = 0; r < 8; ++r) acc[g][n][r] = 0.f;

  float m[2] = {-3.0e38f, -3.0e38f};
  float l[2] = {0.f, 0.f};
  const float L2E = 1.4426950408889634f;

  for (int kt = 0; kt < SEQ / 32; ++kt) {
    const int keyg = b * SEQ + kt * 32;

    // Prefetch next key tile into cache (global_prefetch_b8).
    if (kt + 1 < SEQ / 32) {
      // next K tile: 32 rows, this wave's 128B dk slice; one row per lane.
      __builtin_prefetch(Kh + (size_t)(keyg + 32 + lane) * D_K + dkb, 0, 1);
      // next V tile: 64 dv rows x 64B; two rows per lane.
      __builtin_prefetch(Vt + (size_t)(dkb + lane) * N_ROWS + keyg + 32, 0, 1);
      __builtin_prefetch(Vt + (size_t)(dkb + 32 + lane) * N_ROWS + keyg + 32, 0, 1);
    }

    // --- partial scores: 8 WMMA (2 row-tiles x 2 key-subtiles x 2 k-chunks) ---
    v8f sc[2][2];
#pragma unroll
    for (int g = 0; g < 2; ++g)
#pragma unroll
      for (int t = 0; t < 2; ++t)
#pragma unroll
        for (int r = 0; r < 8; ++r) sc[g][t][r] = 0.f;
#pragma unroll
    for (int t = 0; t < 2; ++t) {
#pragma unroll
      for (int c = 0; c < 2; ++c) {
        F16Frag kf;
        const _Float16* kp = Kh + (size_t)(keyg + t * 16 + ln) * D_K + dkb + c * 32 + 8 * h;
        kf.h8[0] = *(const v8h*)kp;
        kf.h8[1] = *(const v8h*)(kp + 16);
#pragma unroll
        for (int g = 0; g < 2; ++g)          // K fragment reused across row-tiles
          sc[g][t] = wmma_f16f32(qf[g][c], kf, sc[g][t]);
      }
    }
    // store partials (C layout: M = r + 8h, key = 16t + ln)
#pragma unroll
    for (int g = 0; g < 2; ++g)
#pragma unroll
      for (int t = 0; t < 2; ++t)
#pragma unroll
        for (int r = 0; r < 8; ++r)
          S_part[wid][g * 512 + (r + 8 * h) * 32 + t * 16 + ln] = sc[g][t][r];
    __syncthreads();

    // --- deterministic reduction across the 8 dk-slices (float4 vectorized) ---
    {
      const int e4 = threadIdx.x * 4;
      float4 s = {0.f, 0.f, 0.f, 0.f};
#pragma unroll
      for (int w = 0; w < 8; ++w) {
        float4 p = *(const float4*)&S_part[w][e4];
        s.x += p.x; s.y += p.y; s.z += p.z; s.w += p.w;
      }
      *(float4*)&S_red[e4] = s;
    }
    __syncthreads();

    // --- online softmax: lane pair (ln, ln+16) owns row ln of each row-tile ---
    F16Frag pf[2];
#pragma unroll
    for (int g = 0; g < 2; ++g) {
      const float* srow = &S_red[g * 512 + ln * 32];
      float4 a0 = *(const float4*)&srow[8 * h];
      float4 a1 = *(const float4*)&srow[8 * h + 4];
      float4 a2 = *(const float4*)&srow[16 + 8 * h];
      float4 a3 = *(const float4*)&srow[16 + 8 * h + 4];
      float sv[16] = {a0.x, a0.y, a0.z, a0.w, a1.x, a1.y, a1.z, a1.w,
                      a2.x, a2.y, a2.z, a2.w, a3.x, a3.y, a3.z, a3.w};
      float mx = sv[0];
#pragma unroll
      for (int i = 1; i < 16; ++i) mx = fmaxf(mx, sv[i]);
      mx = fmaxf(mx, __shfl_xor(mx, 16, 32));
      const float m_new = fmaxf(m[g], mx);
      const float alpha = exp2f((m[g] - m_new) * L2E);
      float ssum = 0.f;
#pragma unroll
      for (int i = 0; i < 16; ++i) {
        const float p = exp2f((sv[i] - m_new) * L2E);
        sv[i] = p;
        ssum += p;
      }
      ssum += __shfl_xor(ssum, 16, 32);
      l[g] = l[g] * alpha + ssum;
      m[g] = m_new;

      // broadcast per-row alpha into accumulator row layout (M = r + 8h)
      if (lane < 16) bc[wid][g][lane] = alpha;   // wave-internal LDS, in order
      float af[8];
#pragma unroll
      for (int r = 0; r < 8; ++r) af[r] = bc[wid][g][8 * h + r];
#pragma unroll
      for (int n = 0; n < 4; ++n)
#pragma unroll
        for (int r = 0; r < 8; ++r) acc[g][n][r] *= af[r];

      // P already sits in exact A-fragment element order.
#pragma unroll
      for (int i = 0; i < 16; ++i) pf[g].h[i] = (_Float16)sv[i];
    }

    // --- PV: 8 WMMA, V fragment reused across both row-tiles ---
#pragma unroll
    for (int n = 0; n < 4; ++n) {
      F16Frag vf;
      const _Float16* vp = Vt + (size_t)(dkb + n * 16 + ln) * N_ROWS + keyg + 8 * h;
      vf.h8[0] = *(const v8h*)vp;
      vf.h8[1] = *(const v8h*)(vp + 16);
#pragma unroll
      for (int g = 0; g < 2; ++g)
        acc[g][n] = wmma_f16f32(pf[g], vf, acc[g][n]);
    }
    __syncthreads();   // protect S_part / S_red before next iteration
  }

  // normalize by row sums and store
#pragma unroll
  for (int g = 0; g < 2; ++g) {
    if (lane < 16) bc[wid][g][lane] = l[g];
    float lf[8];
#pragma unroll
    for (int r = 0; r < 8; ++r) lf[r] = 1.f / bc[wid][g][8 * h + r];
#pragma unroll
    for (int n = 0; n < 4; ++n)
#pragma unroll
      for (int r = 0; r < 8; ++r)
        out[(size_t)(qbase + g * 16 + r + 8 * h) * D_K + dkb + n * 16 + ln] =
            acc[g][n][r] * lf[r];
  }
}

// ---------------------------------------------------------------------------
extern "C" void kernel_launch(void* const* d_in, const int* in_sizes, int n_in,
                              void* d_out, int out_size, void* d_ws, size_t ws_size,
                              hipStream_t stream) {
  (void)in_sizes; (void)n_in; (void)out_size; (void)ws_size;
  const float* x  = (const float*)d_in[0];
  const float* Wq = (const float*)d_in[1];
  const float* Wk = (const float*)d_in[2];
  const float* Wv = (const float*)d_in[3];
  float* out = (float*)d_out;

  const size_t NQ = (size_t)N_ROWS * D_K;   // 8,388,608 f16 elements
  _Float16* Qh = (_Float16*)d_ws;           // pre-scaled by 1/sqrt(dk)
  _Float16* Kh = Qh + NQ;
  _Float16* Vt = Kh + NQ;                   // Vt is [512][16384]
  _Float16* Wt = Vt + NQ;                   // [3][512][1024]

  // 1) weight convert/transpose
  wconv_kernel<<<dim3((D_K * D_IN) / 256, 3), 256, 0, stream>>>(Wq, Wk, Wv, Wt);
  // 2) QKV projection: 3 * 1024 * 8 strips, 4 waves/block
  qkv_proj_kernel<<<(3 * 1024 * 8) / 4, 128, 0, stream>>>(x, Wt, Qh, Kh, Vt);
  // 3) flash attention: (128 query tiles of 32, 4 batches), 8 waves/block
  attn_kernel<<<dim3(SEQ / 32, 4), 256, 0, stream>>>(Qh, Kh, Vt, out);
}
